// DividedModel_45612552683818
// MI455X (gfx1250) — compile-verified, matrix operand order checked by reading, not run on hardware
//
#include <hip/hip_runtime.h>

// ---------------------------------------------------------------------------
// 64 independent MLP towers, fused into one kernel.
//   h0 = relu(x @ W0[o] + b0[o])      [B,32]x[32,64]
//   h1 = relu(h0 @ W1[o] + b1[o])     [B,64]x[64,64]
//   h2 = relu(h1 @ W2[o] + b2[o])     [B,64]x[64,64]
//   out[:,o] = h2 @ W3[o] + b3[o]     [B,64].[64]
// One block = (tower o, 128 batch rows). All matmuls via
// v_wmma_f32_16x16x32_f16 (f16 operands, f32 accumulate).
// ---------------------------------------------------------------------------

typedef _Float16 v16h __attribute__((ext_vector_type(16)));
typedef _Float16 v8h  __attribute__((ext_vector_type(8)));
typedef _Float16 v4h  __attribute__((ext_vector_type(4)));
typedef float    v8f  __attribute__((ext_vector_type(8)));

constexpr int Bsz = 16384;
constexpr int Isz = 32;
constexpr int Osz = 64;
constexpr int Hsz = 64;
constexpr int TILE_M = 128;   // batch rows per block
constexpr int XS = 40;        // LDS row stride (halves) for K=32 buffers (80B, 16B-aligned)
constexpr int HS = 72;        // LDS row stride (halves) for K=64 buffers (144B, 16B-aligned)

__device__ __forceinline__ v8f wmma_f16(v16h a, v16h b, v8f c) {
  // D = A(16x32 f16) * B(32x16 f16) + C(16x16 f32)
  return __builtin_amdgcn_wmma_f32_16x16x32_f16(
      /*neg_a=*/false, a, /*neg_b=*/false, b,
      /*c_mod=*/(short)0, c, /*reuse_a=*/false, /*reuse_b=*/false);
}

// Single-instruction ReLU (avoid fmaxf's NaN-canonicalization double-max).
__device__ __forceinline__ float relu(float v) { return v > 0.0f ? v : 0.0f; }

// Load one K=32 f16 operand fragment from a row-major LDS tile.
// ISA 16-bit A/B layout (wave32): lane L -> row (L&15), half-group g = L>>4;
//   elements 0..7  = row[koff + g*8 .. +7]
//   elements 8..15 = row[koff + 16 + g*8 .. +7]
// Both are contiguous 16-byte chunks -> two ds_load_b128.
__device__ __forceinline__ v16h ld_frag(const _Float16* base, int row, int stride,
                                        int koff, int g) {
  const _Float16* p = base + row * stride + koff + g * 8;
  v8h lo = *(const v8h*)(p);
  v8h hi = *(const v8h*)(p + 16);
  return __builtin_shufflevector(lo, hi, 0,1,2,3,4,5,6,7,8,9,10,11,12,13,14,15);
}

__global__ __launch_bounds__(256) void towers64_kernel(
    const float* __restrict__ x,
    const float* __restrict__ W0, const float* __restrict__ b0,
    const float* __restrict__ W1, const float* __restrict__ b1,
    const float* __restrict__ W2, const float* __restrict__ b2,
    const float* __restrict__ W3, const float* __restrict__ b3,
    float* __restrict__ out) {
  __shared__ alignas(16) _Float16 sX [TILE_M * XS];  // x tile, f16 [128][32]
  __shared__ alignas(16) _Float16 sH0[TILE_M * HS];  // hidden ping [128][64]
  __shared__ alignas(16) _Float16 sH1[TILE_M * HS];  // hidden pong [128][64]
  __shared__ alignas(16) _Float16 sW0[Hsz * XS];     // W0^T [h][i]
  __shared__ alignas(16) _Float16 sW1[Hsz * HS];     // W1^T [k][h]
  __shared__ alignas(16) _Float16 sW2[Hsz * HS];     // W2^T [k][h]
  __shared__ float sB0[Hsz], sB1[Hsz], sB2[Hsz], sW3[Hsz];

  const int o   = blockIdx.y;           // tower
  const int m0  = blockIdx.x * TILE_M;  // first batch row of tile
  const int tid = threadIdx.x;

  // ---- stage x tile: fp32 -> f16, vectorized float4 loads ----
  {
    const int nvec = TILE_M * Isz / 4;  // 1024
    for (int idx = tid; idx < nvec; idx += 256) {
      const int r  = idx >> 3;
      const int c4 = (idx & 7) * 4;
      const float4 v = *(const float4*)(x + (size_t)(m0 + r) * Isz + c4);
      v4h h; h[0] = (_Float16)v.x; h[1] = (_Float16)v.y;
             h[2] = (_Float16)v.z; h[3] = (_Float16)v.w;
      *(v4h*)(sX + r * XS + c4) = h;
    }
  }
  // ---- stage weights (transposed to [N][K]) + biases ----
  {
    const float* w0g = W0 + (size_t)o * Isz * Hsz;  // [i][h]
    for (int idx = tid; idx < Isz * Hsz; idx += 256) {
      const int i = idx >> 6, h = idx & 63;
      sW0[h * XS + i] = (_Float16)w0g[idx];
    }
    const float* w1g = W1 + (size_t)o * Hsz * Hsz;  // [h][k]
    const float* w2g = W2 + (size_t)o * Hsz * Hsz;
    for (int idx = tid; idx < Hsz * Hsz; idx += 256) {
      const int h = idx >> 6, k = idx & 63;
      sW1[k * HS + h] = (_Float16)w1g[idx];
      sW2[k * HS + h] = (_Float16)w2g[idx];
    }
    if (tid < Hsz) {
      sB0[tid] = b0[o * Hsz + tid];
      sB1[tid] = b1[o * Hsz + tid];
      sB2[tid] = b2[o * Hsz + tid];
      sW3[tid] = W3[o * Hsz + tid];
    }
  }
  __syncthreads();

  const int lane = tid & 31;    // wave32 lane
  const int w    = tid >> 5;    // wave id 0..7; wave owns rows [16w,16w+16)
  const int nl   = lane & 15;
  const int g    = lane >> 4;
  const int rowA = w * 16 + nl; // A-operand row (batch row within tile)

  // ---- layer 0: K = 32, one WMMA per 16x16 tile ----
  {
    const v16h a = ld_frag(sX, rowA, XS, 0, g);
#pragma unroll
    for (int t = 0; t < 4; ++t) {
      const v16h b = ld_frag(sW0, 16 * t + nl, XS, 0, g);
      const float bv = sB0[16 * t + nl];
      v8f c;
#pragma unroll
      for (int e = 0; e < 8; ++e) c[e] = bv;   // C layout: bias depends only on column
      c = wmma_f16(a, b, c);
#pragma unroll
      for (int r = 0; r < 8; ++r)              // D[r+8g][16t+nl], relu -> f16
        sH0[(w * 16 + 8 * g + r) * HS + 16 * t + nl] = (_Float16)relu(c[r]);
    }
  }
  __syncthreads();

  // ---- layer 1: K = 64, two chained WMMAs per tile ----
  {
    const v16h a0 = ld_frag(sH0, rowA, HS, 0,  g);
    const v16h a1 = ld_frag(sH0, rowA, HS, 32, g);
#pragma unroll
    for (int t = 0; t < 4; ++t) {
      const v16h bl = ld_frag(sW1, 16 * t + nl, HS, 0,  g);
      const v16h bh = ld_frag(sW1, 16 * t + nl, HS, 32, g);
      const float bv = sB1[16 * t + nl];
      v8f c;
#pragma unroll
      for (int e = 0; e < 8; ++e) c[e] = bv;
      c = wmma_f16(a0, bl, c);
      c = wmma_f16(a1, bh, c);
#pragma unroll
      for (int r = 0; r < 8; ++r)
        sH1[(w * 16 + 8 * g + r) * HS + 16 * t + nl] = (_Float16)relu(c[r]);
    }
  }
  __syncthreads();

  // ---- layer 2 (WMMA) fused with layer 3 (dot with W3) ----
  float p[8];
#pragma unroll
  for (int r = 0; r < 8; ++r) p[r] = 0.0f;
  {
    const v16h a0 = ld_frag(sH1, rowA, HS, 0,  g);
    const v16h a1 = ld_frag(sH1, rowA, HS, 32, g);
#pragma unroll
    for (int t = 0; t < 4; ++t) {
      const v16h bl = ld_frag(sW2, 16 * t + nl, HS, 0,  g);
      const v16h bh = ld_frag(sW2, 16 * t + nl, HS, 32, g);
      const float bv = sB2[16 * t + nl];
      v8f c;
#pragma unroll
      for (int e = 0; e < 8; ++e) c[e] = bv;
      c = wmma_f16(a0, bl, c);
      c = wmma_f16(a1, bh, c);
      const float w3v = sW3[16 * t + nl];
#pragma unroll
      for (int r = 0; r < 8; ++r)
        p[r] += relu(c[r]) * w3v;              // h2[row][col]*W3[col], col across lanes
    }
  }
  // reduce across the 16 lanes of each half-wave (rows differ between halves)
#pragma unroll
  for (int off = 1; off < 16; off <<= 1) {
#pragma unroll
    for (int r = 0; r < 8; ++r) p[r] += __shfl_xor(p[r], off, 16);
  }
  if (nl == 0) {
    const float bb = b3[o];
    const int row = m0 + w * 16 + 8 * g;       // lane0 -> rows +0..7, lane16 -> rows +8..15
#pragma unroll
    for (int r = 0; r < 8; ++r)                // write-once output: non-temporal stores
      __builtin_nontemporal_store(p[r] + bb, &out[(size_t)(row + r) * Osz + o]);
  }
}

extern "C" void kernel_launch(void* const* d_in, const int* in_sizes, int n_in,
                              void* d_out, int out_size, void* d_ws, size_t ws_size,
                              hipStream_t stream) {
  const float* x  = (const float*)d_in[0];
  const float* W0 = (const float*)d_in[1];
  const float* b0 = (const float*)d_in[2];
  const float* W1 = (const float*)d_in[3];
  const float* b1 = (const float*)d_in[4];
  const float* W2 = (const float*)d_in[5];
  const float* b2 = (const float*)d_in[6];
  const float* W3 = (const float*)d_in[7];
  const float* b3 = (const float*)d_in[8];

  dim3 grid(Bsz / TILE_M, Osz);   // (128, 64) blocks
  towers64_kernel<<<grid, 256, 0, stream>>>(x, W0, b0, W1, b1, W2, b2, W3, b3,
                                            (float*)d_out);
}
